// GNN_27041114096436
// MI455X (gfx1250) — matrix operand order repaired and checked by reading, not compiled, not run on hardware
//
#include <hip/hip_runtime.h>
#include <hip/hip_bf16.h>

#define N_NODES 1024
#define DIM 128
#define LDS_STRIDE 136   // A tiles: 128 halves + 8 pad -> 272B row; staggers banks for b128 reads
#define G_STRIDE 132     // G tile: 128 floats + 4 pad -> 4-bank stagger per row

typedef __attribute__((ext_vector_type(16))) __bf16 v16bf;
typedef __attribute__((ext_vector_type(8)))  __bf16 v8bf;
typedef __attribute__((ext_vector_type(8)))  float  v8f;
typedef __attribute__((ext_vector_type(4)))  float  v4f;
typedef __attribute__((ext_vector_type(8)))  unsigned short us8;

#if __has_builtin(__builtin_amdgcn_tanhf)
#define TANHF(x) __builtin_amdgcn_tanhf(x)
#else
#define TANHF(x) tanhf(x)
#endif

// native RTNE f32 -> bf16 split (hardware cvt instructions)
__device__ __forceinline__ void bf16_split(float x, unsigned short& h, unsigned short& l) {
  __bf16 hb = (__bf16)x;
  float  hf = (float)hb;
  __bf16 lb = (__bf16)(x - hf);
  h = __builtin_bit_cast(unsigned short, hb);
  l = __builtin_bit_cast(unsigned short, lb);
}

__device__ __forceinline__ unsigned lds_off(const void* p) {
  return (unsigned)(uintptr_t)p;   // low 32 bits of generic address = LDS byte offset
}

// ---------------------------------------------------------------------------
// prep: G[m,d] = b_pm[d] + feat[m,:] . W_pm[d, 128:256]
//       R[m,d] = tanh(b_ft[d] + feat[m,:] . W_ft[d, :])
// ---------------------------------------------------------------------------
__global__ __launch_bounds__(DIM) void gnn_prep(
    const float* __restrict__ feat, const float* __restrict__ Wpm,
    const float* __restrict__ bpm,  const float* __restrict__ Wft,
    const float* __restrict__ bft,  float* __restrict__ G, float* __restrict__ R)
{
  const int m = blockIdx.x;
  const int d = threadIdx.x;
  __shared__ float frow[DIM];
  frow[d] = feat[(size_t)m * DIM + d];
  __syncthreads();
  float g = bpm[d];
  float r = bft[d];
  const float* wp = Wpm + (size_t)d * (2 * DIM) + DIM;
  const float* wf = Wft + (size_t)d * DIM;
#pragma unroll 8
  for (int k = 0; k < DIM; ++k) {
    g = fmaf(frow[k], wp[k], g);
    r = fmaf(frow[k], wf[k], r);
  }
  G[(size_t)m * DIM + d] = g;
  R[(size_t)m * DIM + d] = TANHF(r);
}

// ---------------------------------------------------------------------------
// main: per block n, loop 16-row m tiles; bf16x3-split WMMA GEMM vs W_pm[:,:128],
//       G tile staged via async LDS DMA, epilogue +G, tanh, weight by (adj + I).
// ---------------------------------------------------------------------------
__global__ __launch_bounds__(256) void gnn_main(
    const float* __restrict__ pdj, const float* __restrict__ adj,
    const float* __restrict__ Wpm,
    const float* __restrict__ G,   const float* __restrict__ R,
    float* __restrict__ out)
{
  __shared__ __align__(16) unsigned short Ahi[2][16 * LDS_STRIDE];
  __shared__ __align__(16) unsigned short Alo[2][16 * LDS_STRIDE];
  __shared__ __align__(16) float Gld[2][16 * G_STRIDE];
  __shared__ __align__(16) float adw[2][16];

  const int n    = blockIdx.x;
  const int tid  = threadIdx.x;
  const int wave = tid >> 5;
  const int lane = tid & 31;
  const int hgrp = lane >> 4;          // 0: lanes 0-15, 1: lanes 16-31
  const int l15  = lane & 15;
  const int dloc = (wave << 4) + l15;  // this lane's output column d

  // ---- B fragments: W_pm[:, 0:128]^T as bf16 hi/lo, resident in VGPRs ----
  // B 32x16 layout: lane = column d; lanes 0-15 hold K 0..15, lanes 16-31 hold K 16..31.
  v16bf Bhi[4], Blo[4];
  {
    const int koff = hgrp << 4;
#pragma unroll
    for (int kc = 0; kc < 4; ++kc) {
      const float* wrow = Wpm + (size_t)dloc * (2 * DIM) + kc * 32 + koff;
#pragma unroll
      for (int j = 0; j < 16; ++j) {
        float w = wrow[j];
        __bf16 hb = (__bf16)w;
        Bhi[kc][j] = hb;
        Blo[kc][j] = (__bf16)(w - (float)hb);
      }
    }
  }

  const int mrow = tid >> 4;           // staging: row 0..15 of tile
  const int kcol = (tid & 15) << 3;    // staging: 8 contiguous k / d
  const size_t rowbase = (size_t)n * N_NODES * DIM;

  // ---- preload + stage tile 0 into buffer 0 ----
  v4f ra, rb;
  float adval = 0.f;
  {
    const float* src = pdj + rowbase + (size_t)mrow * DIM + kcol;
    ra = *(const v4f*)src;
    rb = *(const v4f*)(src + 4);
    if (tid < 16) adval = adj[(size_t)n * N_NODES + tid] + ((tid == n) ? 1.f : 0.f);

    // async-DMA the G tile (16x128 f32) straight into LDS
    const float* gsrc = G + (size_t)mrow * DIM + kcol;
    unsigned goff = lds_off(&Gld[0][mrow * G_STRIDE + kcol]);
    asm volatile("global_load_async_to_lds_b128 %0, %1, off"
                 :: "v"(goff), "v"(gsrc) : "memory");
    asm volatile("global_load_async_to_lds_b128 %0, %1, off"
                 :: "v"(goff + 16u), "v"(gsrc + 4) : "memory");

    us8 h, l;
#pragma unroll
    for (int i = 0; i < 8; ++i) {
      float x = (i < 4) ? ra[i] : rb[i - 4];
      unsigned short hh, ll;
      bf16_split(x, hh, ll);
      h[i] = hh; l[i] = ll;
    }
    *(us8*)&Ahi[0][mrow * LDS_STRIDE + kcol] = h;
    *(us8*)&Alo[0][mrow * LDS_STRIDE + kcol] = l;
    if (tid < 16) adw[0][tid] = adval;
  }
  asm volatile("s_wait_asynccnt 0" ::: "memory");
  __syncthreads();

  float mfacc = 0.f;
  const int g0 = hgrp << 4;  // A-frag k offset (0 or 16), contiguous 16 k per lane half

  for (int t = 0; t < 64; ++t) {
    const int buf   = t & 1;

    // issue global loads + async G DMA for next tile (overlap with WMMA below)
    if (t + 1 < 64) {
      const float* src = pdj + rowbase + (size_t)((t + 1) * 16 + mrow) * DIM + kcol;
      ra = *(const v4f*)src;
      rb = *(const v4f*)(src + 4);
      if (tid < 16) {
        const int mn = (t + 1) * 16 + tid;
        adval = adj[(size_t)n * N_NODES + mn] + ((mn == n) ? 1.f : 0.f);
      }
      const float* gsrc = G + (size_t)((t + 1) * 16 + mrow) * DIM + kcol;
      unsigned goff = lds_off(&Gld[buf ^ 1][mrow * G_STRIDE + kcol]);
      asm volatile("global_load_async_to_lds_b128 %0, %1, off"
                   :: "v"(goff), "v"(gsrc) : "memory");
      asm volatile("global_load_async_to_lds_b128 %0, %1, off"
                   :: "v"(goff + 16u), "v"(gsrc + 4) : "memory");
      if (t + 2 < 64)
        __builtin_prefetch(pdj + rowbase + (size_t)((t + 2) * 16 + mrow) * DIM + kcol, 0, 0);
    }

    // ---- WMMA: P(16m x 16d) += A(16x128,bf16 hi/lo) x W slice, f32 accum ----
    v8f c = {0.f, 0.f, 0.f, 0.f, 0.f, 0.f, 0.f, 0.f};
#pragma unroll
    for (int kc = 0; kc < 4; ++kc) {
      const int abase = l15 * LDS_STRIDE + kc * 32 + g0;
      const v8bf ah0 = *(const v8bf*)&Ahi[buf][abase];
      const v8bf ah1 = *(const v8bf*)&Ahi[buf][abase + 8];
      const v8bf al0 = *(const v8bf*)&Alo[buf][abase];
      const v8bf al1 = *(const v8bf*)&Alo[buf][abase + 8];
      v16bf ahi, alo;
#pragma unroll
      for (int i = 0; i < 8; ++i) {
        ahi[i] = ah0[i]; ahi[8 + i] = ah1[i];
        alo[i] = al0[i]; alo[8 + i] = al1[i];
      }
      c = __builtin_amdgcn_wmma_f32_16x16x32_bf16(false, ahi, false, Bhi[kc], (short)0, c, false, false);
      c = __builtin_amdgcn_wmma_f32_16x16x32_bf16(false, alo, false, Bhi[kc], (short)0, c, false, false);
      c = __builtin_amdgcn_wmma_f32_16x16x32_bf16(false, ahi, false, Blo[kc], (short)0, c, false, false);
    }

    // ---- epilogue: +G (from LDS), tanh, weight by (adj+I), accumulate over m ----
    const v4f adv0 = *(const v4f*)&adw[buf][hgrp << 3];
    const v4f adv1 = *(const v4f*)&adw[buf][(hgrp << 3) + 4];
#pragma unroll
    for (int r = 0; r < 8; ++r) {
      const int mr = (hgrp << 3) + r;
      float p = c[r] + Gld[buf][mr * G_STRIDE + dloc];
      float f = TANHF(p);
      float w = (r < 4) ? adv0[r] : adv1[r - 4];
      mfacc = fmaf(w, f, mfacc);
    }

    // ---- store staged A tile into the other buffer ----
    if (t + 1 < 64) {
      const int nbuf = buf ^ 1;
      us8 h, l;
#pragma unroll
      for (int i = 0; i < 8; ++i) {
        float x = (i < 4) ? ra[i] : rb[i - 4];
        unsigned short hh, ll;
        bf16_split(x, hh, ll);
        h[i] = hh; l[i] = ll;
      }
      *(us8*)&Ahi[nbuf][mrow * LDS_STRIDE + kcol] = h;
      *(us8*)&Alo[nbuf][mrow * LDS_STRIDE + kcol] = l;
      if (tid < 16) adw[nbuf][tid] = adval;
    }
    asm volatile("s_wait_asynccnt 0" ::: "memory");
    __syncthreads();
  }

  // lanes L and L+16 hold the same d column -> pairwise reduce, add residual, store
  float o = __shfl_xor(mfacc, 16, 32);
  if (hgrp == 0) {
    out[(size_t)n * DIM + dloc] = mfacc + o + R[(size_t)n * DIM + dloc];
  }
}

// ---------------------------------------------------------------------------
extern "C" void kernel_launch(void* const* d_in, const int* in_sizes, int n_in,
                              void* d_out, int out_size, void* d_ws, size_t ws_size,
                              hipStream_t stream) {
  const float* features = (const float*)d_in[0];
  const float* adj      = (const float*)d_in[1];
  const float* pdj      = (const float*)d_in[2];
  const float* W_pm     = (const float*)d_in[3];
  const float* b_pm     = (const float*)d_in[4];
  const float* W_ft     = (const float*)d_in[5];
  const float* b_ft     = (const float*)d_in[6];
  (void)in_sizes; (void)n_in; (void)out_size; (void)ws_size;

  float* G = (float*)d_ws;                      // [1024,128] f32
  float* R = G + (size_t)N_NODES * DIM;         // [1024,128] f32
  float* out = (float*)d_out;

  gnn_prep<<<N_NODES, DIM, 0, stream>>>(features, W_pm, b_pm, W_ft, b_ft, G, R);
  gnn_main<<<N_NODES, 256, 0, stream>>>(pdj, adj, W_pm, G, R, out);
}
